// LinearAttention_63711544869279
// MI455X (gfx1250) — matrix operand order repaired and checked by reading, not compiled, not run on hardware
//
#include <hip/hip_runtime.h>
#include <hip/hip_bf16.h>

// LinearAttention: B=8, L=2048, H=32, E=64, D=64
// Per (b,l): kv = (elu(K)+1)^T @ V / sqrt(L); x = (elu(Q)+1) @ kv * y
// One wave32 per (b,l) pair; bf16 WMMA (v_wmma_f32_16x16x32_bf16), f32 accum.

typedef __attribute__((ext_vector_type(8)))  __bf16 v8bf;
typedef __attribute__((ext_vector_type(16))) __bf16 v16bf;
typedef __attribute__((ext_vector_type(8)))  float  v8f;

#define B_  8
#define L_  2048
#define H_  32
#define E_  64
#define D_  64
#define EPS_ 1e-6f

// LDS row strides (bf16 elements); padded to cut bank conflicts, keep 16B align
#define ST_T 40   // kfT [64][32+8], vT [64][32+8]
#define ST_W 72   // qf  [32][64+8], kvT [64][64+8]

__device__ __forceinline__ float elu1(float x) {
    // elu(x)+1 = x>0 ? x+1 : exp(x)
    return x > 0.f ? x + 1.f : __expf(x);
}

// A fragment (16x32 bf16): lanes 0-15 hold K={0..7,16..23}, lanes 16-31 K={8..15,24..31}
__device__ __forceinline__ v16bf load_A(const __bf16* rowBase, int sel) {
    v8bf lo = *(const v8bf*)(rowBase + 8 * sel);
    v8bf hi = *(const v8bf*)(rowBase + 16 + 8 * sel);
    return __builtin_shufflevector(lo, hi, 0,1,2,3,4,5,6,7, 8,9,10,11,12,13,14,15);
}

// B fragment (32x16 bf16): lane = N, K = {8*sel+0..7, 16+8*sel+0..7} (matches A)
__device__ __forceinline__ v16bf load_B(const __bf16* colBase16) {
    v8bf lo = *(const v8bf*)(colBase16);
    v8bf hi = *(const v8bf*)(colBase16 + 16);
    return __builtin_shufflevector(lo, hi, 0,1,2,3,4,5,6,7, 8,9,10,11,12,13,14,15);
}

__global__ __launch_bounds__(32)
void linattn_wmma_kernel(const float* __restrict__ Q,
                         const float* __restrict__ K,
                         const float* __restrict__ V,
                         float* __restrict__ X)
{
    __shared__ __bf16 kfT[E_ * ST_T];   // [e][h]  (kf transposed)
    __shared__ __bf16 vT [D_ * ST_T];   // [d][h]  (V transposed)
    __shared__ __bf16 qf [H_ * ST_W];   // [h][e]  (row major)
    __shared__ __bf16 kvT[D_ * ST_W];   // [d][e]  (kv transposed)
    __shared__ float  ksumL[E_];
    __shared__ float  yL[H_];

    const int lane = threadIdx.x;       // 0..31
    const int p    = blockIdx.x;        // pair index b*L + l
    const float* qp = Q + (long)p * (H_ * E_);
    const float* kp = K + (long)p * (H_ * E_);
    const float* vp = V + (long)p * (H_ * D_);
    float*       xp = X + (long)p * (H_ * D_);

    const float rsqrtL = rsqrtf((float)L_);

    // ---- stream K: kf = elu+1 -> bf16 kfT[e][h]; ksum = sum over h ----
    float s0 = 0.f, s1 = 0.f;
    for (int h = 0; h < H_; ++h) {
        float2 k2 = *(const float2*)(kp + h * E_ + 2 * lane);
        float f0 = elu1(k2.x), f1 = elu1(k2.y);
        kfT[(2 * lane    ) * ST_T + h] = (__bf16)f0;
        kfT[(2 * lane + 1) * ST_T + h] = (__bf16)f1;
        s0 += f0; s1 += f1;
    }
    ksumL[2 * lane]     = s0;
    ksumL[2 * lane + 1] = s1;

    // ---- stream V -> bf16 vT[d][h] ----
    for (int h = 0; h < H_; ++h) {
        float2 v2 = *(const float2*)(vp + h * D_ + 2 * lane);
        vT[(2 * lane    ) * ST_T + h] = (__bf16)v2.x;
        vT[(2 * lane + 1) * ST_T + h] = (__bf16)v2.y;
    }

    // ---- stream Q: qf = elu+1 -> bf16 qf[h][e] ----
    for (int h = 0; h < H_; ++h) {
        float2 q2 = *(const float2*)(qp + h * E_ + 2 * lane);
        qf[h * ST_W + 2 * lane    ] = (__bf16)elu1(q2.x);
        qf[h * ST_W + 2 * lane + 1] = (__bf16)elu1(q2.y);
    }
    __syncthreads();

    // ---- y[h] = 1 / (qf[h,:] . ksum + eps), f32 accumulate ----
    {
        float acc = EPS_;
        #pragma unroll 8
        for (int e = 0; e < E_; ++e)
            acc += (float)qf[lane * ST_W + e] * ksumL[e];
        yL[lane] = 1.f / acc;
    }

    const int n   = lane & 15;   // N column within tile
    const int sel = lane >> 4;   // half-wave select

    // ---- matmul 1: kv[e][d] = sum_h kfT[e][h] * V[h][d], scaled by L^-1/2 ----
    // M = E (4 tiles), N = D (4 tiles), K = H = 32 (single WMMA k-step)
    for (int mt = 0; mt < 4; ++mt) {
        v16bf a = load_A(&kfT[(16 * mt + n) * ST_T], sel);
        for (int nt = 0; nt < 4; ++nt) {
            v16bf b = load_B(&vT[(16 * nt + n) * ST_T + 8 * sel]);
            v8f c = {};
            c = __builtin_amdgcn_wmma_f32_16x16x32_bf16(
                    false, a, false, b, (short)0, c, false, false);
            // C layout: element r -> M = r + 8*sel (= e_local), N = n (= d_local)
            #pragma unroll
            for (int r = 0; r < 8; ++r) {
                int e = 16 * mt + 8 * sel + r;      // contiguous over r -> b128 store
                int d = 16 * nt + n;
                kvT[d * ST_W + e] = (__bf16)(c[r] * rsqrtL);
            }
        }
    }
    __syncthreads();

    // ---- matmul 2: x[h][d] = (sum_e qf[h][e] * kv[e][d]) * y[h] ----
    // M = H (2 tiles), N = D (4 tiles), K = E = 64 (2 WMMA k-steps)
    for (int mt = 0; mt < 2; ++mt) {
        for (int nt = 0; nt < 4; ++nt) {
            v8f c = {};
            #pragma unroll
            for (int ks = 0; ks < 2; ++ks) {
                v16bf a = load_A(&qf[(16 * mt + n) * ST_W + 32 * ks], sel);
                v16bf b = load_B(&kvT[(16 * nt + n) * ST_W + 32 * ks + 8 * sel]);
                c = __builtin_amdgcn_wmma_f32_16x16x32_bf16(
                        false, a, false, b, (short)0, c, false, false);
            }
            #pragma unroll
            for (int r = 0; r < 8; ++r) {
                int h = 16 * mt + 8 * sel + r;
                int d = 16 * nt + n;                 // contiguous over lanes 0-15
                xp[h * D_ + d] = c[r] * yL[h];
            }
        }
    }
}

extern "C" void kernel_launch(void* const* d_in, const int* in_sizes, int n_in,
                              void* d_out, int out_size, void* d_ws, size_t ws_size,
                              hipStream_t stream) {
    const float* Q = (const float*)d_in[0];
    const float* K = (const float*)d_in[1];
    const float* V = (const float*)d_in[2];
    float*       X = (float*)d_out;

    dim3 grid(B_ * L_);   // 16384 (b,l) pairs
    dim3 block(32);       // one wave32 per pair
    hipLaunchKernelGGL(linattn_wmma_kernel, grid, block, 0, stream, Q, K, V, X);
}
